// SelfAttention_58145267253553
// MI455X (gfx1250) — compile-verified, matrix operand order checked by reading, not run on hardware
//
#include <hip/hip_runtime.h>

// ---------------------------------------------------------------------------
// B=32, C=256, H=64, W=64, NHEADS=8, HEADLEN=32
//   1) qkv = conv3x3(inputs, conv_w)            implicit GEMM, WMMA bf16
//   2) per (b,c): S = Q^T K * scale; O = S V^T  WMMA bf16, LDS-staged
//   3) per-channel GroupNorm fused into (2)
// qkv layout: [3][B][C][H*W] bf16 so each (b,c) attention slice is an 8 KB
// contiguous plane (coalesced staging, async-to-LDS friendly).
// ---------------------------------------------------------------------------

typedef __attribute__((ext_vector_type(16))) __bf16 v16bf;
typedef __attribute__((ext_vector_type(8)))  __bf16 v8bf;
typedef __attribute__((ext_vector_type(8)))  float  v8f;
typedef int v4i __attribute__((vector_size(16)));      // matches builtin param
typedef __attribute__((address_space(1))) v4i* gv4i_p; // global (AS1)
typedef __attribute__((address_space(3))) v4i* lv4i_p; // LDS (AS3)

union ABfrag {
    v16bf v;
    v8bf  h8[2];
};

#define CIN   256
#define COUT  768
#define HH    64
#define WW    64
#define BB    32
#define HW    4096
#define SCALE 0.35355339059327373f   // NHEADS^-0.5
#define EPS   1e-5f
#define LDP   72                     // padded LDS row stride (144 B, 16B-aligned)

#if defined(__has_builtin)
#if __has_builtin(__builtin_amdgcn_global_load_async_to_lds_b128)
#define HAVE_ASYNC_LDS 1
#endif
#if __has_builtin(__builtin_amdgcn_s_wait_asynccnt)
#define HAVE_WAIT_ASYNC 1
#endif
#endif

// ---------------------------------------------------------------------------
// Convert input fp32 [B][C][H][W] -> bf16 channels-last [B][H][W][C]
// ---------------------------------------------------------------------------
__global__ void cvt_input_kernel(const float* __restrict__ in,
                                 __bf16* __restrict__ out) {
    unsigned o = blockIdx.x * 256u + threadIdx.x;   // total 33,554,432
    unsigned c = o & 255u;
    unsigned x = (o >> 8) & 63u;
    unsigned y = (o >> 14) & 63u;
    unsigned b = o >> 20;
    unsigned iidx = (((b << 8) + c) << 12) + (y << 6) + x;
    out[o] = (__bf16)in[iidx];
}

// ---------------------------------------------------------------------------
// Convert weights fp32 [Cout=768][Cin=256][3][3] -> bf16 [tap=9][Cout][Cin]
// ---------------------------------------------------------------------------
__global__ void cvt_weight_kernel(const float* __restrict__ w,
                                  __bf16* __restrict__ out) {
    unsigned o = blockIdx.x * 256u + threadIdx.x;   // total 9*768*256
    unsigned ci = o & 255u;
    unsigned n  = (o >> 8) % 768u;
    unsigned t  = o / (768u * 256u);
    out[o] = (__bf16)w[(n * 256u + ci) * 9u + t];
}

// ---------------------------------------------------------------------------
// Conv 3x3 implicit GEMM: M = B*H*W = 131072, N = 768, K = 9*256.
// One wave: 16(M) x 64(N), K-loop = 9 taps x 8 chunks of 32, 4 f32 accums.
// Output: bf16 qkv planes [3][B][256][4096].
// ---------------------------------------------------------------------------
__global__ void __launch_bounds__(256)
conv_qkv_wmma_kernel(const __bf16* __restrict__ xin,   // [B][H][W][256]
                     const __bf16* __restrict__ wt,    // [9][768][256]
                     __bf16* __restrict__ qkv) {       // [3][B][256][4096]
    const int lane  = threadIdx.x & 31;
    const int gwave = blockIdx.x * 8 + (threadIdx.x >> 5);
    const int mt = gwave / 12;          // m-tile of 16 spatial positions
    const int ng = gwave % 12;          // n-group of 64 output channels

    const int m  = lane & 15;
    const int hi = lane >> 4;
    const int s  = mt * 16 + m;         // global spatial id
    const int b  = s >> 12;
    const int y  = (s >> 6) & 63;
    const int x  = s & 63;

    const v16bf vzero = {};
    v8f acc[4] = {{}, {}, {}, {}};

    for (int t = 0; t < 9; ++t) {
        const int yy = y + (t / 3) - 1;
        const int xx = x + (t % 3) - 1;
        const bool valid = ((unsigned)yy < 64u) && ((unsigned)xx < 64u);
        const __bf16* apix = xin + ((((b * 64 + yy) * 64) + xx) << 8);
        const __bf16* wrow = wt + ((size_t)(t * 768 + ng * 64 + m)) * 256;

        // prefetch next tap's pixel row (global_prefetch_b8)
        if (t < 8) {
            const int yn = y + ((t + 1) / 3) - 1;
            const int xn = x + ((t + 1) % 3) - 1;
            if (((unsigned)yn < 64u) && ((unsigned)xn < 64u))
                __builtin_prefetch(xin + ((((b * 64 + yn) * 64) + xn) << 8), 0, 0);
        }

#pragma unroll
        for (int kc = 0; kc < 8; ++kc) {
            const int kb = kc * 32 + hi * 8;   // lane-half K offset
            ABfrag a;
            if (valid) {
                a.h8[0] = *(const v8bf*)(apix + kb);
                a.h8[1] = *(const v8bf*)(apix + kb + 16);
            } else {
                a.v = vzero;
            }
#pragma unroll
            for (int nt = 0; nt < 4; ++nt) {
                ABfrag bmat;
                const __bf16* wb = wrow + nt * 16 * 256 + kb;
                bmat.h8[0] = *(const v8bf*)(wb);
                bmat.h8[1] = *(const v8bf*)(wb + 16);
                acc[nt] = __builtin_amdgcn_wmma_f32_16x16x32_bf16(
                    false, a.v, false, bmat.v, (short)0, acc[nt], false, false);
            }
        }
    }

    // C/D layout: VGPR r -> row (r + hi*8), col = lane&15
#pragma unroll
    for (int nt = 0; nt < 4; ++nt) {
#pragma unroll
        for (int r = 0; r < 8; ++r) {
            const int ss = mt * 16 + r + hi * 8;
            const int n  = ng * 64 + nt * 16 + m;
            const int qsel = n >> 8;            // 0=Q 1=K 2=V
            const int cc   = n & 255;
            const size_t o = (((size_t)qsel * BB + (ss >> 12)) * 256 + cc) * HW
                           + (ss & 4095);
            qkv[o] = (__bf16)acc[nt][r];
        }
    }
}

// ---------------------------------------------------------------------------
// Attention + fused GroupNorm, one block (128 thr, 4 waves) per (b,c) slice.
// Q,K staged TRANSPOSED into padded LDS so every WMMA fragment (both matmuls,
// A and B operands) is exactly two contiguous ds_load_b128s.
// ---------------------------------------------------------------------------
__global__ void __launch_bounds__(128)
attn_groupnorm_kernel(const __bf16* __restrict__ qkv,  // [3][B][256][4096]
                      const float* __restrict__ gamma,
                      const float* __restrict__ beta,
                      float* __restrict__ out) {       // [B][256][64][64]
    __shared__ __align__(16) __bf16 smem[4 * 64 * LDP];   // 36 KB
    __bf16* sQt = smem;                 // Q transposed: [x][y], stride LDP
    __bf16* sKt = smem + 64 * LDP;      // K transposed: [x][y]
    __bf16* sV  = smem + 2 * 64 * LDP;  // V row-major:  [y][x]
    __bf16* sS  = smem + 3 * 64 * LDP;  // S row-major:  [i][v]
    float*  sO  = (float*)smem;         // 16 KB, reuses sQt+sKt after barrier
    __shared__ float rs[128];
    __shared__ float rq[128];

    const int tid = threadIdx.x;
    const int b = blockIdx.x >> 8;
    const int c = blockIdx.x & 255;

    const __bf16* Qg = qkv + ((size_t)(b * 256 + c)) * HW;
    const __bf16* Kg = Qg + (size_t)BB * 256 * HW;
    const __bf16* Vg = Kg + (size_t)BB * 256 * HW;

    // ---- stage: V row-major (async if available), Q/K transposed ----
    for (int t = tid; t < 512; t += 128) {       // 512 chunks of 8 elements
        const int yy = t >> 3;
        const int x8 = (t & 7) * 8;
#if defined(HAVE_ASYNC_LDS)
        __builtin_amdgcn_global_load_async_to_lds_b128(
            (gv4i_p)(Vg + yy * 64 + x8),
            (lv4i_p)&sV[yy * LDP + x8], 0, 0);
#else
        *(v8bf*)&sV[yy * LDP + x8] = *(const v8bf*)(Vg + yy * 64 + x8);
#endif
        const v8bf q8 = *(const v8bf*)(Qg + yy * 64 + x8);
        const v8bf k8 = *(const v8bf*)(Kg + yy * 64 + x8);
#pragma unroll
        for (int e = 0; e < 8; ++e) {
            sQt[(x8 + e) * LDP + yy] = q8[e];
            sKt[(x8 + e) * LDP + yy] = k8[e];
        }
    }
#if defined(HAVE_ASYNC_LDS)
#if defined(HAVE_WAIT_ASYNC)
    __builtin_amdgcn_s_wait_asynccnt(0);
#else
    asm volatile("s_wait_asynccnt 0x0" ::: "memory");
#endif
#endif
    __syncthreads();

    const int lane = tid & 31;
    const int wv   = tid >> 5;           // 0..3
    const int i0   = wv * 16;            // output row block of this wave
    const int m    = lane & 15;
    const int hi   = lane >> 4;

    // ---- S = Q^T K (contract over y) : 16x64 row block ----
    v8f sacc[4] = {{}, {}, {}, {}};
#pragma unroll
    for (int kc = 0; kc < 2; ++kc) {
        const int kb = kc * 32 + hi * 8;
        ABfrag a;                        // A[m][k] = Q[y=k][i0+m] = sQt[(i0+m)][k]
        a.h8[0] = *(const v8bf*)&sQt[(i0 + m) * LDP + kb];
        a.h8[1] = *(const v8bf*)&sQt[(i0 + m) * LDP + kb + 16];
#pragma unroll
        for (int jt = 0; jt < 4; ++jt) {
            const int j = jt * 16 + m;
            ABfrag bmat;                 // B[k][n] = K[y=k][j] = sKt[j][k]
            bmat.h8[0] = *(const v8bf*)&sKt[j * LDP + kb];
            bmat.h8[1] = *(const v8bf*)&sKt[j * LDP + kb + 16];
            sacc[jt] = __builtin_amdgcn_wmma_f32_16x16x32_bf16(
                false, a.v, false, bmat.v, (short)0, sacc[jt], false, false);
        }
    }
    // scale + store own S row block (only this wave reads it back)
#pragma unroll
    for (int jt = 0; jt < 4; ++jt) {
#pragma unroll
        for (int r = 0; r < 8; ++r) {
            const int i = i0 + r + hi * 8;
            const int j = jt * 16 + m;
            sS[i * LDP + j] = (__bf16)(sacc[jt][r] * SCALE);
        }
    }

    // ---- O = S V^T (contract over v) ----
    v8f oacc[4] = {{}, {}, {}, {}};
#pragma unroll
    for (int kc = 0; kc < 2; ++kc) {
        const int kb = kc * 32 + hi * 8;
        ABfrag a;                        // A[m][k] = S[i0+m][v=k]
        a.h8[0] = *(const v8bf*)&sS[(i0 + m) * LDP + kb];
        a.h8[1] = *(const v8bf*)&sS[(i0 + m) * LDP + kb + 16];
#pragma unroll
        for (int jt = 0; jt < 4; ++jt) {
            const int j = jt * 16 + m;
            ABfrag bmat;                 // B[k][n] = V[j][v=k]
            bmat.h8[0] = *(const v8bf*)&sV[j * LDP + kb];
            bmat.h8[1] = *(const v8bf*)&sV[j * LDP + kb + 16];
            oacc[jt] = __builtin_amdgcn_wmma_f32_16x16x32_bf16(
                false, a.v, false, bmat.v, (short)0, oacc[jt], false, false);
        }
    }

    __syncthreads();   // all waves done with sQt/sKt before reuse as sO
#pragma unroll
    for (int jt = 0; jt < 4; ++jt) {
#pragma unroll
        for (int r = 0; r < 8; ++r) {
            const int i = i0 + r + hi * 8;
            const int j = jt * 16 + m;
            sO[i * 64 + j] = oacc[jt][r];
        }
    }
    __syncthreads();

    // ---- fused per-channel GroupNorm over the 4096 elements ----
    float sum = 0.f, sumsq = 0.f;
    for (int idx = tid; idx < 4096; idx += 128) {
        const float v = sO[idx];
        sum += v;
        sumsq += v * v;
    }
    rs[tid] = sum;
    rq[tid] = sumsq;
    __syncthreads();
    for (int off = 64; off > 0; off >>= 1) {
        if (tid < off) {
            rs[tid] += rs[tid + off];
            rq[tid] += rq[tid + off];
        }
        __syncthreads();
    }
    const float mean = rs[0] * (1.0f / 4096.0f);
    const float var  = rq[0] * (1.0f / 4096.0f) - mean * mean;
    const float rstd = rsqrtf(var + EPS);
    const float g    = gamma[c];
    const float bt   = beta[c];

    float* op = out + ((size_t)(b * 256 + c)) * HW;
    for (int idx = tid; idx < 4096; idx += 128) {
        op[idx] = (sO[idx] - mean) * rstd * g + bt;
    }
}

// ---------------------------------------------------------------------------
// Launcher
// ---------------------------------------------------------------------------
extern "C" void kernel_launch(void* const* d_in, const int* in_sizes, int n_in,
                              void* d_out, int out_size, void* d_ws, size_t ws_size,
                              hipStream_t stream) {
    const float* inputs = (const float*)d_in[0];
    // d_in[1] = time_emb (unused by reference)
    const float* conv_w = (const float*)d_in[2];
    const float* gamma  = (const float*)d_in[3];
    const float* beta   = (const float*)d_in[4];
    float* out = (float*)d_out;

    __bf16* xin  = (__bf16*)d_ws;                       // 33,554,432 elems
    __bf16* wt   = xin + (size_t)BB * HH * WW * CIN;    //  1,769,472 elems
    __bf16* qkvb = wt + (size_t)9 * COUT * CIN;         // 100,663,296 elems

    cvt_input_kernel<<<(BB * CIN * HH * WW) / 256, 256, 0, stream>>>(inputs, xin);
    cvt_weight_kernel<<<(9 * COUT * CIN) / 256, 256, 0, stream>>>(conv_w, wt);

    conv_qkv_wmma_kernel<<<98304 / 8, 256, 0, stream>>>(xin, wt, qkvb);

    attn_groupnorm_kernel<<<BB * CIN, 128, 0, stream>>>(qkvb, gamma, beta, out);
}
